// CombinedModel_Av2_44092134261185
// MI455X (gfx1250) — compile-verified, matrix operand order checked by reading, not run on hardware
//
#include <hip/hip_runtime.h>
#include <hip/hip_bf16.h>

typedef __attribute__((ext_vector_type(16))) _Float16 v16h;
typedef __attribute__((ext_vector_type(8)))  _Float16 v8h;
typedef __attribute__((ext_vector_type(8)))  float    v8f;

#define CDIV(a,b) (((a)+(b)-1)/(b))

// ---- model dims ----
#define B_       4
#define SAMPLES  64000
#define HOP      160
#define NFFT     2048
#define PADW     1024
#define NFR      401
#define MROWS    1604        // B_*NFR
#define MPAD     1632        // pad to /32 (gemm tile)
#define NBINS    1025
#define NBINSP   1056        // pad to /32
#define NMELS    229
#define CCH      48
#define F1W      57
#define F2W      14
#define DM       256
#define FF       1024
#define NH       8
#define DHD      32
#define NCLS     39
#define NCLSP    64          // pad to /32
#define KPROJ    672         // 48*14 (21*32)
#define TT       8
#define DROWS    32          // 4*8 decoder rows
#define NC_TOK   39

// Async global->LDS copy of 16B per lane (CDNA5 TDM-lite path, ASYNCcnt).
__device__ __forceinline__ void async_b128_to_lds(const _Float16* g, _Float16* l)
{
  unsigned int lo = (unsigned int)(unsigned long long)l;   // LDS byte offset
  asm volatile("global_load_async_to_lds_b128 %0, %1, off"
               :: "v"(lo), "v"(g) : "memory");
}

// ============================================================
// WMMA GEMM: C[M x N](f32) = A[M x K](f16,row-major) * B[K x N](f16,row-major)
// 128 threads (4 waves) -> 32x32 C tile. Double-buffered LDS tiles filled by
// GLOBAL_LOAD_ASYNC_TO_LDS_B128, consumption gated by s_wait_asynccnt.
// M,N multiples of 32; K multiple of 32. Optional bias + relu.
// ============================================================
__global__ __launch_bounds__(128)
void k_gemm_wmma(const _Float16* __restrict__ A,
                 const _Float16* __restrict__ Bm,
                 const float* __restrict__ bias,
                 float* __restrict__ C,
                 int K, int lda, int ldb, int ldc,
                 int realN, int relu)
{
  __shared__ __attribute__((aligned(16))) _Float16 As[2][32 * 32];
  __shared__ __attribute__((aligned(16))) _Float16 Bs[2][32 * 32];

  const int tid  = threadIdx.x;
  const int wave = tid >> 5;            // 0..3
  const int lane = tid & 31;
  const int l15  = lane & 15;
  const int half = lane >> 4;
  const int wm   = (wave >> 1) * 16;    // sub-tile row within 32x32
  const int wn   = (wave & 1) * 16;     // sub-tile col within 32x32
  const int n0   = blockIdx.x * 32;
  const int m0   = blockIdx.y * 32;

  // cooperative tile fetch: 128 threads x 16B = 2KB (32x32 f16) per matrix
  const int lrow = tid >> 2;            // 0..31
  const int lcol = (tid & 3) * 8;       // 0,8,16,24
  const _Float16* gA = A  + (size_t)(m0 + lrow) * lda + lcol;   // advance by kk
  const _Float16* gB = Bm + (size_t)lrow * ldb + n0 + lcol;     // advance by kk*ldb

  v8f acc = (v8f){0.f,0.f,0.f,0.f,0.f,0.f,0.f,0.f};

  // prologue: async-fetch tile 0 into buffer 0
  async_b128_to_lds(gA, &As[0][lrow * 32 + lcol]);
  async_b128_to_lds(gB, &Bs[0][lrow * 32 + lcol]);

  for (int kk = 0, it = 0; kk < K; kk += 32, ++it) {
    const int cur = it & 1;
    if (kk + 32 < K) {
      // overlap: fetch tile it+1 into the other buffer while computing tile it
      async_b128_to_lds(gA + (kk + 32), &As[cur ^ 1][lrow * 32 + lcol]);
      async_b128_to_lds(gB + (size_t)(kk + 32) * ldb, &Bs[cur ^ 1][lrow * 32 + lcol]);
      asm volatile("s_wait_asynccnt 0x2" ::: "memory");   // tile it landed
    } else {
      asm volatile("s_wait_asynccnt 0x0" ::: "memory");
    }
    __syncthreads();                                      // all waves see tile it

    // A fragment: two contiguous 8-half runs -> 2x ds_load_b128
    const _Float16* arow = &As[cur][(wm + l15) * 32];
    v8h alo = *(const v8h*)&arow[(half << 3)];            // k = half*8 + 0..7
    v8h ahi = *(const v8h*)&arow[16 + (half << 3)];       // k = 16+half*8 + 0..7
    v16h af, bf;
#pragma unroll
    for (int j = 0; j < 8; ++j) { af[j] = alo[j]; af[8 + j] = ahi[j]; }
#pragma unroll
    for (int v = 0; v < 8; ++v) {
      const int kb = ((v < 4) ? (v << 1) : (16 + ((v - 4) << 1))) + (half << 3);
      bf[2*v]   = Bs[cur][(size_t)kb * 32 + wn + l15];
      bf[2*v+1] = Bs[cur][(size_t)(kb + 1) * 32 + wn + l15];
    }
    acc = __builtin_amdgcn_wmma_f32_16x16x32_f16(false, af, false, bf,
                                                 (short)0, acc, false, false);
    __syncthreads();   // all waves done reading buffer `cur` (reused at it+2)
  }

  const int col = n0 + wn + l15;
  float bv = (bias != nullptr && col < realN) ? bias[col] : 0.f;
#pragma unroll
  for (int r = 0; r < 8; ++r) {
    const int row = m0 + wm + r + half * 8; // C layout: vgpr r -> M=r / r+8
    float v = acc[r] + bv;
    if (relu) v = fmaxf(v, 0.f);
    C[(size_t)row * ldc + col] = v;         // padded buffers: no guard needed
  }
}

// ============================================================
// STFT front-end
// ============================================================
__global__ void k_frames(const float* __restrict__ wav, const float* __restrict__ win,
                         _Float16* __restrict__ A)
{
  int idx = blockIdx.x * blockDim.x + threadIdx.x;
  if (idx >= MPAD * NFFT) return;
  int n = idx % NFFT, r = idx / NFFT;
  float v = 0.f;
  if (r < MROWS) {
    int b = r / NFR, t = r % NFR;
    int j = t * HOP + n - PADW;            // reflect pad
    if (j < 0) j = -j;
    if (j >= SAMPLES) j = 2 * SAMPLES - 2 - j;
    v = wav[(size_t)b * SAMPLES + j] * win[n];
  }
  A[idx] = (_Float16)v;
}

__global__ void k_dft(_Float16* __restrict__ Dc, _Float16* __restrict__ Ds)
{
  int idx = blockIdx.x * blockDim.x + threadIdx.x;
  if (idx >= NFFT * NBINSP) return;
  int k = idx % NBINSP, n = idx / NBINSP;
  float c = 0.f, s = 0.f;
  if (k < NBINS) {
    int m = (n * k) & (NFFT - 1);          // exploit periodicity for precision
    float ang = -6.283185307179586f * (float)m / (float)NFFT;
    c = cosf(ang); s = sinf(ang);
  }
  Dc[idx] = (_Float16)c;
  Ds[idx] = (_Float16)s;
}

__global__ void k_power(const float* __restrict__ Re, const float* __restrict__ Im,
                        float* __restrict__ P)
{
  int idx = blockIdx.x * blockDim.x + threadIdx.x;
  if (idx >= MROWS * NBINS) return;
  int k = idx % NBINS, r = idx / NBINS;
  float re = Re[(size_t)r * NBINSP + k];
  float im = Im[(size_t)r * NBINSP + k];
  P[idx] = re * re + im * im;
}

__global__ void k_mel(const float* __restrict__ P, const float* __restrict__ fb,
                      float* __restrict__ lm, float* __restrict__ spec)
{
  int idx = blockIdx.x * blockDim.x + threadIdx.x;
  if (idx >= MROWS * NMELS) return;
  int m = idx % NMELS, r = idx / NMELS;
  const float* pr = P + (size_t)r * NBINS;
  float s = 0.f;
  for (int k = 0; k < NBINS; ++k) s += pr[k] * fb[(size_t)k * NMELS + m];
  float x = logf(s + 1e-10f);
  lm[idx] = x;
  spec[idx] = x;
}

// ============================================================
// Conv backbone (conv 3x3 SAME + relu fused with width-maxpool/4)
// ============================================================
__global__ void k_conv1pool(const float* __restrict__ x, const float* __restrict__ w,
                            const float* __restrict__ bias, float* __restrict__ o)
{
  int idx = blockIdx.x * blockDim.x + threadIdx.x;
  if (idx >= B_ * CCH * NFR * F1W) return;
  int fo = idx % F1W, t = (idx / F1W) % NFR;
  int c = (idx / (F1W * NFR)) % CCH, b = idx / (F1W * NFR * CCH);
  float mx = -1e30f;
  for (int fi = 0; fi < 4; ++fi) {
    int f = fo * 4 + fi;
    float acc = bias[c];
    for (int dt = -1; dt <= 1; ++dt) {
      int tt = t + dt; if (tt < 0 || tt >= NFR) continue;
      for (int df = -1; df <= 1; ++df) {
        int ff = f + df; if (ff < 0 || ff >= NMELS) continue;
        acc += x[((size_t)b * NFR + tt) * NMELS + ff] * w[c * 9 + (dt + 1) * 3 + (df + 1)];
      }
    }
    acc = fmaxf(acc, 0.f);
    mx = fmaxf(mx, acc);
  }
  o[idx] = mx;
}

__global__ void k_conv2pool(const float* __restrict__ x, const float* __restrict__ w,
                            const float* __restrict__ bias,
                            float* __restrict__ o, float* __restrict__ o2)
{
  int idx = blockIdx.x * blockDim.x + threadIdx.x;
  if (idx >= B_ * CCH * NFR * F2W) return;
  int fo = idx % F2W, t = (idx / F2W) % NFR;
  int c = (idx / (F2W * NFR)) % CCH, b = idx / (F2W * NFR * CCH);
  float mx = -1e30f;
  for (int fi = 0; fi < 4; ++fi) {
    int f = fo * 4 + fi;
    float acc = bias[c];
    for (int ci = 0; ci < CCH; ++ci) {
      const float* xc = x + (((size_t)b * CCH + ci) * NFR) * F1W;
      const float* wc = w + ((size_t)c * CCH + ci) * 9;
      for (int dt = -1; dt <= 1; ++dt) {
        int tt = t + dt; if (tt < 0 || tt >= NFR) continue;
        for (int df = -1; df <= 1; ++df) {
          int ff = f + df; if (ff < 0 || ff >= F1W) continue;
          acc += xc[(size_t)tt * F1W + ff] * wc[(dt + 1) * 3 + (df + 1)];
        }
      }
    }
    acc = fmaxf(acc, 0.f);
    mx = fmaxf(mx, acc);
  }
  o[idx] = mx;
  o2[idx] = mx;
}

__global__ void k_aproj(const float* __restrict__ bb, _Float16* __restrict__ A)
{
  int idx = blockIdx.x * blockDim.x + threadIdx.x;
  if (idx >= MPAD * KPROJ) return;
  int cf = idx % KPROJ, r = idx / KPROJ;
  float v = 0.f;
  if (r < MROWS) {
    int b = r / NFR, t = r % NFR, c = cf / F2W, f = cf % F2W;
    v = bb[(((size_t)b * CCH + c) * NFR + t) * F2W + f];
  }
  A[idx] = (_Float16)v;
}

__global__ void k_addpos(float* __restrict__ h)
{
  int idx = blockIdx.x * blockDim.x + threadIdx.x;
  if (idx >= MROWS * DM) return;
  int d = idx % DM, r = idx / DM, t = r % NFR;
  int i = d >> 1;
  float div = expf(-(float)(2 * i) * 9.210340371976184f / (float)DM);
  float ang = (float)t * div;
  h[idx] += (d & 1) ? cosf(ang) : sinf(ang);
}

// ============================================================
// Generic helpers
// ============================================================
__global__ void k_f32tof16(const float* __restrict__ s, _Float16* __restrict__ d,
                           int R, int C, int Cpad, int Rpad)
{
  int idx = blockIdx.x * blockDim.x + threadIdx.x;
  if (idx >= Rpad * Cpad) return;
  int c = idx % Cpad, r = idx / Cpad;
  float v = (r < R && c < C) ? s[(size_t)r * C + c] : 0.f;
  d[idx] = (_Float16)v;
}

__global__ void k_add_ln(const float* __restrict__ a, const float* __restrict__ s,
                         const float* __restrict__ g, const float* __restrict__ be,
                         float* __restrict__ o, int rows)
{
  int r = blockIdx.x * blockDim.x + threadIdx.x;
  if (r >= rows) return;
  const float* pa = a + (size_t)r * DM;
  const float* ps = s + (size_t)r * DM;
  float* po = o + (size_t)r * DM;
  float m = 0.f;
  for (int i = 0; i < DM; ++i) m += pa[i] + ps[i];
  m *= (1.f / DM);
  float v = 0.f;
  for (int i = 0; i < DM; ++i) { float d = pa[i] + ps[i] - m; v += d * d; }
  v *= (1.f / DM);
  float inv = rsqrtf(v + 1e-5f);
  for (int i = 0; i < DM; ++i) {
    float d = (pa[i] + ps[i] - m) * inv;
    po[i] = d * g[i] + be[i];
  }
}

__global__ void k_scores(const float* __restrict__ Q, const float* __restrict__ Km,
                         float* __restrict__ S, int QS, int KS, int causal)
{
  int idx = blockIdx.x * blockDim.x + threadIdx.x;
  if (idx >= B_ * NH * QS * KS) return;
  int k = idx % KS, q = (idx / KS) % QS;
  int h = (idx / (KS * QS)) % NH, b = idx / (KS * QS * NH);
  const float* qp = Q + ((size_t)(b * QS + q)) * DM + h * DHD;
  const float* kp = Km + ((size_t)(b * KS + k)) * DM + h * DHD;
  float s = 0.f;
  for (int d = 0; d < DHD; ++d) s += qp[d] * kp[d];
  s *= 0.17677669529663687f;               // 1/sqrt(32)
  if (causal && k > q) s = -1e9f;
  S[idx] = s;
}

__global__ void k_softmax(float* __restrict__ S, int rows, int len)
{
  int r = blockIdx.x * blockDim.x + threadIdx.x;
  if (r >= rows) return;
  float* p = S + (size_t)r * len;
  float mx = -1e30f;
  for (int i = 0; i < len; ++i) mx = fmaxf(mx, p[i]);
  float sum = 0.f;
  for (int i = 0; i < len; ++i) { float e = expf(p[i] - mx); p[i] = e; sum += e; }
  float inv = 1.f / sum;
  for (int i = 0; i < len; ++i) p[i] *= inv;
}

__global__ void k_av(const float* __restrict__ S, const float* __restrict__ V,
                     float* __restrict__ O, int QS, int KS)
{
  int idx = blockIdx.x * blockDim.x + threadIdx.x;
  if (idx >= B_ * NH * QS * DHD) return;
  int d = idx % DHD, q = (idx / DHD) % QS;
  int h = (idx / (DHD * QS)) % NH, b = idx / (DHD * QS * NH);
  const float* sp = S + (((size_t)(b * NH + h) * QS + q)) * KS;
  float acc = 0.f;
  for (int k = 0; k < KS; ++k)
    acc += sp[k] * V[((size_t)(b * KS + k)) * DM + h * DHD + d];
  O[((size_t)(b * QS + q)) * DM + h * DHD + d] = acc;
}

// ============================================================
// Decoder specifics
// ============================================================
__global__ void k_seqinit(int* __restrict__ seq)
{
  int b = blockIdx.x * blockDim.x + threadIdx.x;
  if (b < B_) seq[b * (TT + 1)] = NC_TOK;
}

__global__ void k_embed(const int* __restrict__ seq, const float* __restrict__ qe,
                        float* __restrict__ y, int slen)
{
  int idx = blockIdx.x * blockDim.x + threadIdx.x;
  if (idx >= DROWS * DM) return;
  int d = idx % DM, r = idx / DM, b = r / TT, q = r % TT;
  float v = 0.f;
  if (q < slen) v = qe[(size_t)seq[b * (TT + 1) + q] * DM + d];
  y[idx] = v;
}

__global__ void k_argmax(const float* __restrict__ gl, int* __restrict__ seq, int slen)
{
  int b = blockIdx.x * blockDim.x + threadIdx.x;
  if (b >= B_) return;
  const float* p = gl + ((size_t)(b * TT + slen - 1)) * NCLSP;
  int best = 0; float bv = p[0];
  for (int c = 1; c < NCLS; ++c) if (p[c] > bv) { bv = p[c]; best = c; }
  if (slen <= TT) seq[b * (TT + 1) + slen] = best;
}

__global__ void k_logits_copy(const float* __restrict__ gl, float* __restrict__ o, int slen)
{
  int idx = blockIdx.x * blockDim.x + threadIdx.x;
  if (idx >= B_ * NCLS) return;
  int c = idx % NCLS, b = idx / NCLS;
  o[((size_t)b * TT + (slen - 1)) * NCLS + c] =
      gl[((size_t)(b * TT + slen - 1)) * NCLSP + c];
}

// ============================================================
// Host orchestration
// ============================================================
extern "C" void kernel_launch(void* const* d_in, const int* in_sizes, int n_in,
                              void* d_out, int out_size, void* d_ws, size_t ws_size,
                              hipStream_t stream)
{
  const float* wav     = (const float*)d_in[0];
  const float* mel_fb  = (const float*)d_in[2];
  const float* window  = (const float*)d_in[3];
  const float* c1w     = (const float*)d_in[4];
  const float* c1b     = (const float*)d_in[5];
  const float* c2w     = (const float*)d_in[6];
  const float* c2b     = (const float*)d_in[7];
  const float* projw   = (const float*)d_in[8];
  const float* projb   = (const float*)d_in[9];
  const float* qembed  = (const float*)d_in[10];
  const float* clsw    = (const float*)d_in[11];
  const float* clsb    = (const float*)d_in[12];
  const float* eaw     = (const float*)d_in[13];
  const float* eab     = (const float*)d_in[14];
  const float* eln     = (const float*)d_in[15];
  const float* ef1     = (const float*)d_in[16];
  const float* ef1b    = (const float*)d_in[17];
  const float* ef2     = (const float*)d_in[18];
  const float* ef2b    = (const float*)d_in[19];
  const float* dsaw    = (const float*)d_in[20];
  const float* dsab    = (const float*)d_in[21];
  const float* dcaw    = (const float*)d_in[22];
  const float* dcab    = (const float*)d_in[23];
  const float* dln     = (const float*)d_in[24];
  const float* df1     = (const float*)d_in[25];
  const float* df1b    = (const float*)d_in[26];
  const float* df2     = (const float*)d_in[27];
  const float* df2b    = (const float*)d_in[28];

  float* out_logits = (float*)d_out;                         // 4*8*39
  float* out_spec   = out_logits + B_ * TT * NCLS;           // 4*1*401*229
  float* out_bb     = out_spec + (size_t)B_ * NFR * NMELS;   // 4*48*401*14

  // ---- workspace carve ----
  char* base = (char*)d_ws; size_t off = 0;
  auto alloc = [&](size_t bytes) -> void* {
    void* p = base + off; off += (bytes + 255) & ~(size_t)255; return p;
  };
  _Float16* hAfr  = (_Float16*)alloc((size_t)MPAD * NFFT * 2);
  _Float16* hDc   = (_Float16*)alloc((size_t)NFFT * NBINSP * 2);
  _Float16* hDs   = (_Float16*)alloc((size_t)NFFT * NBINSP * 2);
  float*    fRe   = (float*)alloc((size_t)MPAD * NBINSP * 4);
  float*    fIm   = (float*)alloc((size_t)MPAD * NBINSP * 4);
  float*    fP    = (float*)alloc((size_t)MROWS * NBINS * 4);
  float*    fLM   = (float*)alloc((size_t)MROWS * NMELS * 4);
  float*    fP1   = (float*)alloc((size_t)B_ * CCH * NFR * F1W * 4);
  float*    fBB   = (float*)alloc((size_t)B_ * CCH * NFR * F2W * 4);
  _Float16* hAproj= (_Float16*)alloc((size_t)MPAD * KPROJ * 2);
  float*    fX    = (float*)alloc((size_t)MPAD * DM * 4);
  _Float16* hX16  = (_Float16*)alloc((size_t)MPAD * DM * 2);
  _Float16* hA16  = (_Float16*)alloc((size_t)MPAD * DM * 2);
  float*    fQ    = (float*)alloc((size_t)MPAD * DM * 4);
  float*    fK    = (float*)alloc((size_t)MPAD * DM * 4);
  float*    fV    = (float*)alloc((size_t)MPAD * DM * 4);
  float*    fAo   = (float*)alloc((size_t)MPAD * DM * 4);
  float*    fT    = (float*)alloc((size_t)MPAD * DM * 4);
  float*    fS    = (float*)alloc((size_t)B_ * NH * NFR * NFR * 4);
  float*    fF    = (float*)alloc((size_t)MPAD * FF * 4);
  _Float16* hF16  = (_Float16*)alloc((size_t)MPAD * FF * 2);
  float*    fKc[2], *fVc[2];
  for (int l = 0; l < 2; ++l) {
    fKc[l] = (float*)alloc((size_t)MPAD * DM * 4);
    fVc[l] = (float*)alloc((size_t)MPAD * DM * 4);
  }
  // f16 weights
  _Float16* hProjW = (_Float16*)alloc((size_t)KPROJ * DM * 2);
  _Float16* hEncA  = (_Float16*)alloc((size_t)2 * 4 * DM * DM * 2);
  _Float16* hEncF1 = (_Float16*)alloc((size_t)2 * DM * FF * 2);
  _Float16* hEncF2 = (_Float16*)alloc((size_t)2 * FF * DM * 2);
  _Float16* hDecSA = (_Float16*)alloc((size_t)2 * 4 * DM * DM * 2);
  _Float16* hDecCA = (_Float16*)alloc((size_t)2 * 4 * DM * DM * 2);
  _Float16* hDecF1 = (_Float16*)alloc((size_t)2 * DM * FF * 2);
  _Float16* hDecF2 = (_Float16*)alloc((size_t)2 * FF * DM * 2);
  _Float16* hCls   = (_Float16*)alloc((size_t)DM * NCLSP * 2);
  // decoder buffers
  float*    fY    = (float*)alloc((size_t)DROWS * DM * 4);
  _Float16* hY16  = (_Float16*)alloc((size_t)DROWS * DM * 2);
  _Float16* hAd16 = (_Float16*)alloc((size_t)DROWS * DM * 2);
  float*    fQd   = (float*)alloc((size_t)DROWS * DM * 4);
  float*    fKd   = (float*)alloc((size_t)DROWS * DM * 4);
  float*    fVd   = (float*)alloc((size_t)DROWS * DM * 4);
  float*    fAd   = (float*)alloc((size_t)DROWS * DM * 4);
  float*    fTd   = (float*)alloc((size_t)DROWS * DM * 4);
  float*    fSd   = (float*)alloc((size_t)B_ * NH * TT * TT * 4);
  float*    fS2   = (float*)alloc((size_t)B_ * NH * TT * NFR * 4);
  float*    fFd   = (float*)alloc((size_t)DROWS * FF * 4);
  _Float16* hFd16 = (_Float16*)alloc((size_t)DROWS * FF * 2);
  float*    fGl   = (float*)alloc((size_t)DROWS * NCLSP * 4);
  int*      iSeq  = (int*)alloc((size_t)B_ * (TT + 1) * 4);
  (void)ws_size; (void)in_sizes; (void)n_in; (void)out_size;

  auto cvt = [&](const float* s, _Float16* d, int R, int C, int Cpad, int Rpad) {
    int n = Rpad * Cpad;
    k_f32tof16<<<CDIV(n, 256), 256, 0, stream>>>(s, d, R, C, Cpad, Rpad);
  };
  auto gemm = [&](const _Float16* A, const _Float16* Bm, const float* bias, float* C,
                  int Mp, int Np, int K, int realN, int relu) {
    dim3 g(Np / 32, Mp / 32);
    k_gemm_wmma<<<g, 128, 0, stream>>>(A, Bm, bias, C, K, K, Np, Np, realN, relu);
  };

  // ---- weight conversions (f32 -> f16) ----
  cvt(projw, hProjW, KPROJ, DM, DM, KPROJ);
  for (int l = 0; l < 2; ++l)
    for (int i = 0; i < 4; ++i) {
      cvt(eaw  + (size_t)(l * 4 + i) * DM * DM, hEncA  + (size_t)(l * 4 + i) * DM * DM, DM, DM, DM, DM);
      cvt(dsaw + (size_t)(l * 4 + i) * DM * DM, hDecSA + (size_t)(l * 4 + i) * DM * DM, DM, DM, DM, DM);
      cvt(dcaw + (size_t)(l * 4 + i) * DM * DM, hDecCA + (size_t)(l * 4 + i) * DM * DM, DM, DM, DM, DM);
    }
  for (int l = 0; l < 2; ++l) {
    cvt(ef1 + (size_t)l * DM * FF, hEncF1 + (size_t)l * DM * FF, DM, FF, FF, DM);
    cvt(ef2 + (size_t)l * FF * DM, hEncF2 + (size_t)l * FF * DM, FF, DM, DM, FF);
    cvt(df1 + (size_t)l * DM * FF, hDecF1 + (size_t)l * DM * FF, DM, FF, FF, DM);
    cvt(df2 + (size_t)l * FF * DM, hDecF2 + (size_t)l * FF * DM, FF, DM, DM, FF);
  }
  cvt(clsw, hCls, DM, NCLS, NCLSP, DM);

  // ---- STFT as WMMA GEMM ----
  k_frames<<<CDIV(MPAD * NFFT, 256), 256, 0, stream>>>(wav, window, hAfr);
  k_dft<<<CDIV(NFFT * NBINSP, 256), 256, 0, stream>>>(hDc, hDs);
  gemm(hAfr, hDc, nullptr, fRe, MPAD, NBINSP, NFFT, NBINSP, 0);
  gemm(hAfr, hDs, nullptr, fIm, MPAD, NBINSP, NFFT, NBINSP, 0);
  k_power<<<CDIV(MROWS * NBINS, 256), 256, 0, stream>>>(fRe, fIm, fP);
  k_mel<<<CDIV(MROWS * NMELS, 256), 256, 0, stream>>>(fP, mel_fb, fLM, out_spec);

  // ---- conv backbone ----
  k_conv1pool<<<CDIV(B_ * CCH * NFR * F1W, 256), 256, 0, stream>>>(fLM, c1w, c1b, fP1);
  k_conv2pool<<<CDIV(B_ * CCH * NFR * F2W, 256), 256, 0, stream>>>(fP1, c2w, c2b, fBB, out_bb);

  // ---- projection + positional ----
  k_aproj<<<CDIV(MPAD * KPROJ, 256), 256, 0, stream>>>(fBB, hAproj);
  gemm(hAproj, hProjW, projb, fX, MPAD, DM, KPROJ, DM, 0);
  k_addpos<<<CDIV(MROWS * DM, 256), 256, 0, stream>>>(fX);

  // ---- encoder (2 layers) ----
  for (int l = 0; l < 2; ++l) {
    const _Float16* Wq = hEncA + (size_t)(l * 4 + 0) * DM * DM;
    const _Float16* Wk = hEncA + (size_t)(l * 4 + 1) * DM * DM;
    const _Float16* Wv = hEncA + (size_t)(l * 4 + 2) * DM * DM;
    const _Float16* Wo = hEncA + (size_t)(l * 4 + 3) * DM * DM;
    const float* bq = eab + (size_t)(l * 4 + 0) * DM;
    const float* bk = eab + (size_t)(l * 4 + 1) * DM;
    const float* bv = eab + (size_t)(l * 4 + 2) * DM;
    const float* bo = eab + (size_t)(l * 4 + 3) * DM;
    cvt(fX, hX16, MROWS, DM, DM, MPAD);
    gemm(hX16, Wq, bq, fQ, MPAD, DM, DM, DM, 0);
    gemm(hX16, Wk, bk, fK, MPAD, DM, DM, DM, 0);
    gemm(hX16, Wv, bv, fV, MPAD, DM, DM, DM, 0);
    k_scores<<<CDIV(B_ * NH * NFR * NFR, 256), 256, 0, stream>>>(fQ, fK, fS, NFR, NFR, 0);
    k_softmax<<<CDIV(B_ * NH * NFR, 256), 256, 0, stream>>>(fS, B_ * NH * NFR, NFR);
    k_av<<<CDIV(B_ * NH * NFR * DHD, 256), 256, 0, stream>>>(fS, fV, fAo, NFR, NFR);
    cvt(fAo, hA16, MROWS, DM, DM, MPAD);
    gemm(hA16, Wo, bo, fT, MPAD, DM, DM, DM, 0);
    k_add_ln<<<CDIV(MROWS, 64), 64, 0, stream>>>(fX, fT,
        eln + ((size_t)(l * 2 + 0) * 2 + 0) * DM, eln + ((size_t)(l * 2 + 0) * 2 + 1) * DM,
        fX, MROWS);
    cvt(fX, hX16, MROWS, DM, DM, MPAD);
    gemm(hX16, hEncF1 + (size_t)l * DM * FF, ef1b + (size_t)l * FF, fF, MPAD, FF, DM, FF, 1);
    cvt(fF, hF16, MROWS, FF, FF, MPAD);
    gemm(hF16, hEncF2 + (size_t)l * FF * DM, ef2b + (size_t)l * DM, fT, MPAD, DM, FF, DM, 0);
    k_add_ln<<<CDIV(MROWS, 64), 64, 0, stream>>>(fX, fT,
        eln + ((size_t)(l * 2 + 1) * 2 + 0) * DM, eln + ((size_t)(l * 2 + 1) * 2 + 1) * DM,
        fX, MROWS);
  }

  // ---- decoder cross-attn K/V from encoder memory (hoisted) ----
  cvt(fX, hX16, MROWS, DM, DM, MPAD);
  for (int l = 0; l < 2; ++l) {
    gemm(hX16, hDecCA + (size_t)(l * 4 + 1) * DM * DM, dcab + (size_t)(l * 4 + 1) * DM,
         fKc[l], MPAD, DM, DM, DM, 0);
    gemm(hX16, hDecCA + (size_t)(l * 4 + 2) * DM * DM, dcab + (size_t)(l * 4 + 2) * DM,
         fVc[l], MPAD, DM, DM, DM, 0);
  }

  // ---- autoregressive decode ----
  k_seqinit<<<1, 32, 0, stream>>>(iSeq);
  for (int s = 1; s <= TT; ++s) {
    k_embed<<<CDIV(DROWS * DM, 256), 256, 0, stream>>>(iSeq, qembed, fY, s);
    for (int l = 0; l < 2; ++l) {
      // self-attn (causal)
      cvt(fY, hY16, DROWS, DM, DM, DROWS);
      gemm(hY16, hDecSA + (size_t)(l * 4 + 0) * DM * DM, dsab + (size_t)(l * 4 + 0) * DM, fQd, DROWS, DM, DM, DM, 0);
      gemm(hY16, hDecSA + (size_t)(l * 4 + 1) * DM * DM, dsab + (size_t)(l * 4 + 1) * DM, fKd, DROWS, DM, DM, DM, 0);
      gemm(hY16, hDecSA + (size_t)(l * 4 + 2) * DM * DM, dsab + (size_t)(l * 4 + 2) * DM, fVd, DROWS, DM, DM, DM, 0);
      k_scores<<<CDIV(B_ * NH * TT * TT, 256), 256, 0, stream>>>(fQd, fKd, fSd, TT, TT, 1);
      k_softmax<<<CDIV(B_ * NH * TT, 256), 256, 0, stream>>>(fSd, B_ * NH * TT, TT);
      k_av<<<CDIV(B_ * NH * TT * DHD, 256), 256, 0, stream>>>(fSd, fVd, fAd, TT, TT);
      cvt(fAd, hAd16, DROWS, DM, DM, DROWS);
      gemm(hAd16, hDecSA + (size_t)(l * 4 + 3) * DM * DM, dsab + (size_t)(l * 4 + 3) * DM, fTd, DROWS, DM, DM, DM, 0);
      k_add_ln<<<1, 64, 0, stream>>>(fY, fTd,
          dln + ((size_t)(l * 3 + 0) * 2 + 0) * DM, dln + ((size_t)(l * 3 + 0) * 2 + 1) * DM,
          fY, DROWS);
      // cross-attn
      cvt(fY, hY16, DROWS, DM, DM, DROWS);
      gemm(hY16, hDecCA + (size_t)(l * 4 + 0) * DM * DM, dcab + (size_t)(l * 4 + 0) * DM, fQd, DROWS, DM, DM, DM, 0);
      k_scores<<<CDIV(B_ * NH * TT * NFR, 256), 256, 0, stream>>>(fQd, fKc[l], fS2, TT, NFR, 0);
      k_softmax<<<CDIV(B_ * NH * TT, 256), 256, 0, stream>>>(fS2, B_ * NH * TT, NFR);
      k_av<<<CDIV(B_ * NH * TT * DHD, 256), 256, 0, stream>>>(fS2, fVc[l], fAd, TT, NFR);
      cvt(fAd, hAd16, DROWS, DM, DM, DROWS);
      gemm(hAd16, hDecCA + (size_t)(l * 4 + 3) * DM * DM, dcab + (size_t)(l * 4 + 3) * DM, fTd, DROWS, DM, DM, DM, 0);
      k_add_ln<<<1, 64, 0, stream>>>(fY, fTd,
          dln + ((size_t)(l * 3 + 1) * 2 + 0) * DM, dln + ((size_t)(l * 3 + 1) * 2 + 1) * DM,
          fY, DROWS);
      // FFN
      cvt(fY, hY16, DROWS, DM, DM, DROWS);
      gemm(hY16, hDecF1 + (size_t)l * DM * FF, df1b + (size_t)l * FF, fFd, DROWS, FF, DM, FF, 1);
      cvt(fFd, hFd16, DROWS, FF, FF, DROWS);
      gemm(hFd16, hDecF2 + (size_t)l * FF * DM, df2b + (size_t)l * DM, fTd, DROWS, DM, FF, DM, 0);
      k_add_ln<<<1, 64, 0, stream>>>(fY, fTd,
          dln + ((size_t)(l * 3 + 2) * 2 + 0) * DM, dln + ((size_t)(l * 3 + 2) * 2 + 1) * DM,
          fY, DROWS);
    }
    // classifier + greedy step
    cvt(fY, hY16, DROWS, DM, DM, DROWS);
    gemm(hY16, hCls, clsb, fGl, DROWS, NCLSP, DM, NCLS, 0);
    k_logits_copy<<<CDIV(B_ * NCLS, 128), 128, 0, stream>>>(fGl, out_logits, s);
    k_argmax<<<1, 32, 0, stream>>>(fGl, iSeq, s);
  }
}
